// FastRCNN_17446157157179
// MI455X (gfx1250) — matrix operand order repaired and checked by reading, not compiled, not run on hardware
//
#include <hip/hip_runtime.h>

typedef __bf16 bf16_t;
typedef __attribute__((ext_vector_type(16))) __bf16 v16bf;
typedef __attribute__((ext_vector_type(8)))  __bf16 v8bf;
typedef __attribute__((ext_vector_type(8)))  float   v8f;
typedef __attribute__((ext_vector_type(4)))  unsigned int v4u;
typedef __attribute__((ext_vector_type(4)))  int     v4i;
typedef __attribute__((ext_vector_type(8)))  int     v8i;

#define P_TOT   100352      // 2048 samples * 49 positions
#define NSAMP   2048
#define CCH     256
#define HID     12544       // 256*49
#define REG_OUT 204
#define CLS_OUT 51
#define NFIN    256         // 204 reg + 51 cls + 1 zero pad

union FragU { v16bf v; v8bf h[2]; };

__device__ __forceinline__ v8f wmma_bf16(v16bf a, v16bf b, v8f c) {
    return __builtin_amdgcn_wmma_f32_16x16x32_bf16(false, a, false, b, (short)0, c,
                                                   false, false);
}

// A fragment: 16(M=rows) x 32(K), rows are positions/samples, row-major ld=K
__device__ __forceinline__ v16bf load_a(const bf16_t* __restrict__ base, int ld,
                                        int m0, int lane, int k0) {
    const bf16_t* row = base + (size_t)(m0 + (lane & 15)) * ld + k0;
    const int kh = (lane >> 4) << 3;            // 0 or 8
    FragU f;
    f.h[0] = *(const v8bf*)(row + kh);          // K = k0+kh .. +7
    f.h[1] = *(const v8bf*)(row + 16 + kh);     // K = k0+16+kh .. +7
    return f.v;
}

// B fragment: 32(K) x 16(N=out channels), from W[Cout][Cin] row-major ld=K
__device__ __forceinline__ v16bf load_b(const bf16_t* __restrict__ base, int ld,
                                        int n0, int lane, int k0) {
    const bf16_t* row = base + (size_t)(n0 + (lane & 15)) * ld + k0 + ((lane >> 4) << 3);
    FragU f;
    f.h[0] = *(const v8bf*)(row);
    f.h[1] = *(const v8bf*)(row + 16);
    return f.v;
}

__device__ __forceinline__ float silu_f(float x) { return x / (1.f + __expf(-x)); }

// ---------------------------------------------------------------------------
// TDM: DMA a [rows x chunk] bf16 slab (row stride ld elements) from global
// into LDS at byte offset lds_off, packed as [rows][chunk]. D# built per
// CDNA5 ISA (2D tile, data_size=2B, type=2).
// 6-arg builtin form (clang-23 / therock headers).
// ---------------------------------------------------------------------------
__device__ __forceinline__ void tdm_load_slab(const bf16_t* gsrc, unsigned lds_off,
                                              int rows, int ld, int chunk) {
    const unsigned long long ga = (unsigned long long)(uintptr_t)gsrc;
    v4u g0;
    g0[0] = 1u;                                           // count=1, user mode
    g0[1] = lds_off;                                      // lds_addr (bytes)
    g0[2] = (unsigned)(ga & 0xFFFFFFFFu);                 // global_addr[31:0]
    g0[3] = ((unsigned)(ga >> 32) & 0x01FFFFFFu) | 0x80000000u;  // addr[56:32] | type=2
    v8i g1;
    g1[0] = 0x00010000;                                   // wg_mask=0, data_size=1 (2B)
    g1[1] = (int)(((unsigned)ld & 0xFFFFu) << 16);        // tensor_dim0[15:0]
    g1[2] = (int)(((unsigned)ld >> 16) |
                  (((unsigned)rows & 0xFFFFu) << 16));    // tensor_dim0 hi | tensor_dim1 lo
    g1[3] = (int)(((unsigned)rows >> 16) |
                  (((unsigned)chunk & 0xFFFFu) << 16));   // tensor_dim1 hi | tile_dim0
    g1[4] = rows;                                         // tile_dim1 (tile_dim2=0)
    g1[5] = ld;                                           // tensor_dim0_stride[31:0]
    g1[6] = 0;                                            // stride hi | dim1_stride lo
    g1[7] = 0;
    const v4i z4 = {0, 0, 0, 0};                          // groups 2/3 unused (2D)
    const v8i z8 = {0, 0, 0, 0, 0, 0, 0, 0};
    __builtin_amdgcn_tensor_load_to_lds(g0, g1, z4, z4, z8, 0);
}

// ---------- pe_proj GEMM + bias + SiLU -> S (bf16 [P][256]) ----------
// 4 waves/block, each wave owns 4 N-tiles (shares one A fragment across them).
__global__ void k_gemm_silu(const bf16_t* __restrict__ A, const bf16_t* __restrict__ W,
                            const float* __restrict__ bias, bf16_t* __restrict__ Out) {
    constexpr int K = 256;
    const int lane = threadIdx.x & 31;
    const int wave = threadIdx.x >> 5;
    const int m0 = blockIdx.x << 4;
    v8f acc[4];
#pragma unroll
    for (int t = 0; t < 4; ++t) acc[t] = (v8f){0.f,0.f,0.f,0.f,0.f,0.f,0.f,0.f};
#pragma unroll
    for (int k = 0; k < K; k += 32) {
        v16bf a = load_a(A, K, m0, lane, k);
#pragma unroll
        for (int t = 0; t < 4; ++t) {
            v16bf b = load_b(W, K, (wave * 4 + t) << 4, lane, k);
            acc[t] = wmma_bf16(a, b, acc[t]);
        }
    }
    const int mb = m0 + ((lane >> 4) << 3);
#pragma unroll
    for (int t = 0; t < 4; ++t) {
        const int n = ((wave * 4 + t) << 4) + (lane & 15);
        const float bz = bias[n];
#pragma unroll
        for (int i = 0; i < 8; ++i)
            Out[(size_t)(mb + i) * CCH + n] = (bf16_t)silu_f(acc[t][i] + bz);
    }
}

// ---------- bpe GEMM (mul & add halves) + FiLM, in-place on H ----------
// Each wave owns 2 channel tiles, each with its mul/add pair (4 accumulators).
__global__ void k_gemm_film(const bf16_t* __restrict__ S, const bf16_t* __restrict__ Wb,
                            const float* __restrict__ bias, bf16_t* __restrict__ H) {
    constexpr int K = 256;
    const int lane = threadIdx.x & 31;
    const int wave = threadIdx.x >> 5;
    const int m0 = blockIdx.x << 4;
    const int nb = (blockIdx.y * 8 + wave * 2) << 4;      // first of 2 tiles, in [0,1024)
    v8f am[2], aa[2];
#pragma unroll
    for (int t = 0; t < 2; ++t) {
        am[t] = (v8f){0.f,0.f,0.f,0.f,0.f,0.f,0.f,0.f};
        aa[t] = (v8f){0.f,0.f,0.f,0.f,0.f,0.f,0.f,0.f};
    }
#pragma unroll
    for (int k = 0; k < K; k += 32) {
        v16bf a = load_a(S, K, m0, lane, k);
#pragma unroll
        for (int t = 0; t < 2; ++t) {
            v16bf bm = load_b(Wb, K, nb + 16 * t, lane, k);
            v16bf ba = load_b(Wb, K, nb + 16 * t + 1024, lane, k);
            am[t] = wmma_bf16(a, bm, am[t]);
            aa[t] = wmma_bf16(a, ba, aa[t]);
        }
    }
    const int mb = m0 + ((lane >> 4) << 3);
#pragma unroll
    for (int t = 0; t < 2; ++t) {
        const int n = nb + 16 * t + (lane & 15);
        const float bmul = bias[n];
        const float badd = bias[n + 1024];
#pragma unroll
        for (int i = 0; i < 8; ++i) {
            const size_t idx = (size_t)(mb + i) * 1024 + n;
            const float h = (float)H[idx];
            H[idx] = (bf16_t)(h * (am[t][i] + bmul) + (aa[t][i] + badd));
        }
    }
}

// ---------- pt GEMM + bias + BN + SiLU -> X (TDM-staged weights) ----------
// Block covers full N=256 (4 waves x 4 tiles). Weights DMA'd chunk-wise into
// LDS by the TDM; waves consume B fragments via ds_load_b128.
__global__ void k_gemm_bn(const bf16_t* __restrict__ A, const bf16_t* __restrict__ W,
                          const float* __restrict__ bias, const float* __restrict__ g,
                          const float* __restrict__ be, const float* __restrict__ mu,
                          const float* __restrict__ va, bf16_t* __restrict__ X) {
    constexpr int K = 1024, CH = 128;
    __shared__ bf16_t Wl[256 * CH];                       // 64 KB slab
    const int lane = threadIdx.x & 31;
    const int wave = threadIdx.x >> 5;
    const int m0 = blockIdx.x << 4;
    const unsigned lds_base = (unsigned)(uintptr_t)(&Wl[0]);
    v8f acc[4];
#pragma unroll
    for (int t = 0; t < 4; ++t) acc[t] = (v8f){0.f,0.f,0.f,0.f,0.f,0.f,0.f,0.f};
    for (int k0 = 0; k0 < K; k0 += CH) {
        if (k0) __syncthreads();                          // LDS reuse guard
        if (wave == 0) {
            tdm_load_slab(W + k0, lds_base, 256, K, CH);
            __builtin_amdgcn_s_wait_tensorcnt(0);
        }
        __syncthreads();
#pragma unroll
        for (int kk = 0; kk < CH; kk += 32) {
            v16bf a = load_a(A, K, m0, lane, k0 + kk);
#pragma unroll
            for (int t = 0; t < 4; ++t) {
                v16bf b = load_b(Wl, CH, (wave * 4 + t) << 4, lane, kk);
                acc[t] = wmma_bf16(a, b, acc[t]);
            }
        }
    }
    const int mb = m0 + ((lane >> 4) << 3);
#pragma unroll
    for (int t = 0; t < 4; ++t) {
        const int n = ((wave * 4 + t) << 4) + (lane & 15);
        const float bz = bias[n];
        const float sc = g[n] * rsqrtf(va[n] + 1e-5f);
        const float sh = be[n] - mu[n] * sc;
#pragma unroll
        for (int i = 0; i < 8; ++i) {
            float v = (acc[t][i] + bz) * sc + sh;
            X[(size_t)(mb + i) * CCH + n] = (bf16_t)silu_f(v);
        }
    }
}

// ---------- final heads GEMM (TDM-staged weights) -> logits fp32 ----------
__global__ void k_gemm_final(const bf16_t* __restrict__ A, const bf16_t* __restrict__ W,
                             const float* __restrict__ bias, float* __restrict__ L) {
    constexpr int K = HID, CH = 128;                      // 98 chunks
    __shared__ bf16_t Wl[256 * CH];                       // 64 KB slab
    const int lane = threadIdx.x & 31;
    const int wave = threadIdx.x >> 5;
    const int m0 = blockIdx.x << 4;
    const unsigned lds_base = (unsigned)(uintptr_t)(&Wl[0]);
    v8f acc[4];
#pragma unroll
    for (int t = 0; t < 4; ++t) acc[t] = (v8f){0.f,0.f,0.f,0.f,0.f,0.f,0.f,0.f};
    for (int k0 = 0; k0 < K; k0 += CH) {
        if (k0) __syncthreads();
        if (wave == 0) {
            tdm_load_slab(W + k0, lds_base, 256, K, CH);
            __builtin_amdgcn_s_wait_tensorcnt(0);
        }
        __syncthreads();
        // prefetch next A chunk while consuming this one
        __builtin_prefetch(A + (size_t)(m0 + (lane & 15)) * K + k0 + CH, 0, 0);
#pragma unroll
        for (int kk = 0; kk < CH; kk += 32) {
            v16bf a = load_a(A, K, m0, lane, k0 + kk);
#pragma unroll
            for (int t = 0; t < 4; ++t) {
                v16bf b = load_b(Wl, CH, (wave * 4 + t) << 4, lane, kk);
                acc[t] = wmma_bf16(a, b, acc[t]);
            }
        }
    }
    const int mb = m0 + ((lane >> 4) << 3);
#pragma unroll
    for (int t = 0; t < 4; ++t) {
        const int n = ((wave * 4 + t) << 4) + (lane & 15);
        const float bz = bias[n];
#pragma unroll
        for (int i = 0; i < 8; ++i)
            L[(size_t)(mb + i) * NFIN + n] = acc[t][i] + bz;
    }
}

// ---------- depthwise 3x3, groups=256, EXP=4 -> H [P][1024] bf16 ----------
__global__ void k_depthwise(const bf16_t* __restrict__ X, const float* __restrict__ ws9,
                            const float* __restrict__ bs, bf16_t* __restrict__ H) {
    __shared__ bf16_t t[HID];                             // [p*256 + c]
    const int n = blockIdx.x;
    const bf16_t* src = X + (size_t)n * HID;
    for (int i = threadIdx.x; i < HID; i += 256) t[i] = src[i];
    __syncthreads();
#pragma unroll
    for (int j = 0; j < 4; ++j) {
        const int oc = threadIdx.x + (j << 8);            // [0,1024)
        const int gch = oc >> 2;                          // input channel
        float w[9];
#pragma unroll
        for (int q = 0; q < 9; ++q) w[q] = ws9[oc * 9 + q];
        const float bo = bs[oc];
        for (int p = 0; p < 49; ++p) {
            const int y = p / 7, x = p - y * 7;
            float acc = bo;
#pragma unroll
            for (int ky = 0; ky < 3; ++ky) {
                const int yy = y + ky - 1;
                if (yy < 0 || yy > 6) continue;
#pragma unroll
                for (int kx = 0; kx < 3; ++kx) {
                    const int xx = x + kx - 1;
                    if (xx < 0 || xx > 6) continue;
                    acc += w[ky * 3 + kx] * (float)t[(yy * 7 + xx) * CCH + gch];
                }
            }
            H[((size_t)n * 49 + p) * 1024 + oc] = (bf16_t)acc;
        }
    }
}

// ---------- [N,C,49] fp32 -> [N*49][C] bf16 (LDS-staged transpose) ----------
__global__ void k_transpose(const float* __restrict__ src, bf16_t* __restrict__ dst) {
    __shared__ bf16_t t[HID];                             // [c*49 + p]
    const int n = blockIdx.x;
    const float* s = src + (size_t)n * HID;
    for (int i = threadIdx.x; i < HID; i += 256) t[i] = (bf16_t)s[i];
    __syncthreads();
    bf16_t* d = dst + (size_t)n * HID;
    for (int i = threadIdx.x; i < HID; i += 256) {
        const int p = i >> 8, c = i & 255;
        d[i] = t[c * 49 + p];
    }
}

// ---------- generic fp32 -> bf16 ----------
__global__ void k_f2bf(const float* __restrict__ s, bf16_t* __restrict__ d, int n) {
    const int i = blockIdx.x * 256 + threadIdx.x;
    if (i < n) d[i] = (bf16_t)s[i];
}

// ---------- build combined head weight, k' = p*256+c permutation ----------
__global__ void k_build_final(const float* __restrict__ regw, const float* __restrict__ clsw,
                              const float* __restrict__ regb, const float* __restrict__ clsb,
                              bf16_t* __restrict__ Wfin, float* __restrict__ bfin) {
    const int i = blockIdx.x * 256 + threadIdx.x;         // i = o*12544 + k'
    if (i < NFIN * HID) {
        const int o = i / HID;
        const int kk = i - o * HID;
        const int p = kk >> 8, c = kk & 255;
        const int korig = c * 49 + p;
        float v = 0.f;
        if (o < REG_OUT)                v = regw[(size_t)o * HID + korig];
        else if (o < REG_OUT + CLS_OUT) v = clsw[(size_t)(o - REG_OUT) * HID + korig];
        Wfin[i] = (bf16_t)v;
    }
    if (i < NFIN) {
        float b = 0.f;
        if (i < REG_OUT) b = regb[i];
        else if (i < REG_OUT + CLS_OUT) b = clsb[i - REG_OUT];
        bfin[i] = b;
    }
}

// ---------- split logits: copy reg part, softmax cls part ----------
__global__ void k_output(const float* __restrict__ L, float* __restrict__ out) {
    __shared__ float sm[CLS_OUT];
    __shared__ float red[2];
    const int n = blockIdx.x;
    const float* row = L + (size_t)n * NFIN;
    for (int i = threadIdx.x; i < REG_OUT; i += 64)
        out[(size_t)n * REG_OUT + i] = row[i];
    if (threadIdx.x < CLS_OUT) sm[threadIdx.x] = row[REG_OUT + threadIdx.x];
    __syncthreads();
    if (threadIdx.x == 0) {
        float m = -3.0e38f;
        for (int i = 0; i < CLS_OUT; ++i) m = fmaxf(m, sm[i]);
        float s = 0.f;
        for (int i = 0; i < CLS_OUT; ++i) s += __expf(sm[i] - m);
        red[0] = m;
        red[1] = 1.f / s;
    }
    __syncthreads();
    if (threadIdx.x < CLS_OUT) {
        out[(size_t)NSAMP * REG_OUT + (size_t)n * CLS_OUT + threadIdx.x] =
            __expf(sm[threadIdx.x] - red[0]) * red[1];
    }
}

extern "C" void kernel_launch(void* const* d_in, const int* in_sizes, int n_in,
                              void* d_out, int out_size, void* d_ws, size_t ws_size,
                              hipStream_t stream) {
    (void)in_sizes; (void)n_in; (void)out_size; (void)ws_size;
    const float* roi_pool = (const float*)d_in[0];
    const float* roi_pe   = (const float*)d_in[1];
    const float* pe_w     = (const float*)d_in[2];
    const float* pe_b     = (const float*)d_in[3];
    const float* dw_w     = (const float*)d_in[4];
    const float* dw_b     = (const float*)d_in[5];
    const float* bpe_w    = (const float*)d_in[6];
    const float* bpe_b    = (const float*)d_in[7];
    const float* pt_w     = (const float*)d_in[8];
    const float* pt_b     = (const float*)d_in[9];
    const float* bn_g     = (const float*)d_in[10];
    const float* bn_b     = (const float*)d_in[11];
    const float* bn_m     = (const float*)d_in[12];
    const float* bn_v     = (const float*)d_in[13];
    const float* reg_w    = (const float*)d_in[14];
    const float* reg_b    = (const float*)d_in[15];
    const float* cls_w    = (const float*)d_in[16];
    const float* cls_b    = (const float*)d_in[17];

    char* ws = (char*)d_ws;
    size_t off = 0;
    auto take = [&](size_t bytes) -> char* {
        char* p = ws + off;
        off += (bytes + 255) & ~(size_t)255;
        return p;
    };
    bf16_t* Wpe  = (bf16_t*)take((size_t)65536 * 2);
    bf16_t* Wbpe = (bf16_t*)take((size_t)1048576 * 2);     // both depths
    bf16_t* Wpt  = (bf16_t*)take((size_t)524288 * 2);      // both depths
    bf16_t* Wfin = (bf16_t*)take((size_t)NFIN * HID * 2);
    float*  bfin = (float*) take((size_t)NFIN * 4);
    bf16_t* Xpe  = (bf16_t*)take((size_t)P_TOT * CCH * 2);
    bf16_t* X    = (bf16_t*)take((size_t)P_TOT * CCH * 2);
    bf16_t* S    = (bf16_t*)take((size_t)P_TOT * CCH * 2);
    bf16_t* Hb   = (bf16_t*)take((size_t)P_TOT * 1024 * 2);
    float*  Lg   = (float*) take((size_t)NSAMP * NFIN * 4);

    // weight conversion / permutation
    k_f2bf<<<(65536 + 255) / 256, 256, 0, stream>>>(pe_w, Wpe, 65536);
    k_f2bf<<<(1048576 + 255) / 256, 256, 0, stream>>>(bpe_w, Wbpe, 1048576);
    k_f2bf<<<(524288 + 255) / 256, 256, 0, stream>>>(pt_w, Wpt, 524288);
    k_build_final<<<HID, 256, 0, stream>>>(reg_w, cls_w, reg_b, cls_b, Wfin, bfin);

    // activations to position-major bf16
    k_transpose<<<NSAMP, 256, 0, stream>>>(roi_pe, Xpe);
    k_transpose<<<NSAMP, 256, 0, stream>>>(roi_pool, X);

    // S = silu(pe_proj @ pe + b)
    k_gemm_silu<<<dim3(P_TOT / 16), 128, 0, stream>>>(Xpe, Wpe, pe_b, S);

    for (int d = 0; d < 2; ++d) {
        k_depthwise<<<NSAMP, 256, 0, stream>>>(X, dw_w + d * 9216, dw_b + d * 1024, Hb);
        k_gemm_film<<<dim3(P_TOT / 16, 8), 128, 0, stream>>>(
            S, Wbpe + (size_t)d * 524288, bpe_b + d * 2048, Hb);
        k_gemm_bn<<<dim3(P_TOT / 16), 128, 0, stream>>>(
            Hb, Wpt + (size_t)d * 262144, pt_b + d * 256,
            bn_g + d * 256, bn_b + d * 256, bn_m + d * 256, bn_v + d * 256, X);
    }

    k_gemm_final<<<dim3(NSAMP / 16), 128, 0, stream>>>(X, Wfin, bfin, Lg);
    k_output<<<NSAMP, 64, 0, stream>>>(Lg, (float*)d_out);
}
